// MultiHeadAttention_25683904430755
// MI455X (gfx1250) — matrix operand order repaired and checked by reading, not compiled, not run on hardware
//
#include <hip/hip_runtime.h>

typedef __attribute__((ext_vector_type(16))) _Float16 v16h;
typedef __attribute__((ext_vector_type(8)))  float    v8f;
typedef __attribute__((ext_vector_type(4)))  int      v4i;

namespace {
constexpr int kB  = 4;
constexpr int kL  = 2048;
constexpr int kD  = 1024;
constexpr int kH  = 16;
constexpr int kDh = 64;
constexpr int kBL = kB * kL;          // 8192 tokens
constexpr float kLnEps = 1e-5f;
constexpr float kLog10000 = 9.210340371976184f;
}

// ---------------------------------------------------------------------------
// Async global->LDS staging (gfx1250 GLOBAL_LOAD_ASYNC_TO_LDS_B128, ASYNCcnt).
// Signature (from probe): (v4i AS1*, v4i AS3*, imm offset, imm cpol).
// ---------------------------------------------------------------------------
#if __has_builtin(__builtin_amdgcn_global_load_async_to_lds_b128)
#define HAS_ASYNC_LDS 1
#else
#define HAS_ASYNC_LDS 0
#endif

__device__ inline void stage16(const _Float16* g, _Float16* l) {
#if HAS_ASYNC_LDS
  __builtin_amdgcn_global_load_async_to_lds_b128(
      (__attribute__((address_space(1))) v4i*)g,
      (__attribute__((address_space(3))) v4i*)l, 0, 0);
#else
  *(uint4*)l = *(const uint4*)g;
#endif
}
__device__ inline void stage_wait() {
#if HAS_ASYNC_LDS
  asm volatile("s_wait_asynccnt 0x0" ::: "memory");
#endif
}

// ---------------------------------------------------------------------------
// WMMA fragment loaders (wave32 layouts per CDNA5 ISA 7.12.2).
// All loads are two 16-B chunks -> ds_load_b128 / global_load_b128.
// ---------------------------------------------------------------------------
__device__ inline v16h load_frag_a(const _Float16* __restrict__ base, int stride) {
  const int lane = threadIdx.x & 31;
  const int m  = lane & 15;
  const int ko = (lane >> 4) << 3;          // 0 or 8
  const _Float16* p = base + (size_t)m * stride + ko;
  union { uint4 u[2]; v16h v; } r;
  r.u[0] = *(const uint4*)(p);              // K = ko .. ko+7
  r.u[1] = *(const uint4*)(p + 16);         // K = 16+ko .. 16+ko+7
  return r.v;
}

__device__ inline v16h load_frag_b(const _Float16* __restrict__ base, int stride) {
  const int lane = threadIdx.x & 31;
  const int n  = lane & 15;
  const int kb = (lane >> 4) << 3;          // 0 or 8
  const _Float16* p = base + (size_t)n * stride + kb;
  union { uint4 u[2]; v16h v; } r;
  r.u[0] = *(const uint4*)(p);
  r.u[1] = *(const uint4*)(p + 16);
  return r.v;
}

// ---------------------------------------------------------------------------
// Block reduction helper (4 packed sums)
// ---------------------------------------------------------------------------
__device__ inline float4 block_reduce4(float4 v, float4* red) {
  const int lane = threadIdx.x & 31, wave = threadIdx.x >> 5;
#pragma unroll
  for (int m = 1; m < 32; m <<= 1) {
    v.x += __shfl_xor(v.x, m, 32);
    v.y += __shfl_xor(v.y, m, 32);
    v.z += __shfl_xor(v.z, m, 32);
    v.w += __shfl_xor(v.w, m, 32);
  }
  if (lane == 0) red[wave] = v;
  __syncthreads();
  if (threadIdx.x == 0) {
    float4 s = red[0];
    for (int i = 1; i < 8; ++i) {
      s.x += red[i].x; s.y += red[i].y; s.z += red[i].z; s.w += red[i].w;
    }
    red[0] = s;
  }
  __syncthreads();
  return red[0];
}

// ---------------------------------------------------------------------------
// Kernel 1: LayerNorm(x) with weight+bias -> f16
// ---------------------------------------------------------------------------
__global__ __launch_bounds__(256) void mha_ln_x_kernel(
    const float* __restrict__ x, const float* __restrict__ w,
    const float* __restrict__ bias, _Float16* __restrict__ xn) {
  __shared__ float  row[kD];
  __shared__ float4 red[8];
  const int token = blockIdx.x;
  const float* src = x + (size_t)token * kD;
  float s = 0.f, ss = 0.f;
  for (int d = threadIdx.x; d < kD; d += 256) {
    const float v = src[d];
    row[d] = v; s += v; ss += v * v;
  }
  const float4 tot = block_reduce4(make_float4(s, ss, 0.f, 0.f), red);
  const float mu = tot.x * (1.0f / kD);
  const float var = tot.y * (1.0f / kD) - mu * mu;
  const float rs = rsqrtf(var + kLnEps);
  __syncthreads();
  for (int d = threadIdx.x; d < kD; d += 256)
    xn[(size_t)token * kD + d] = (_Float16)((row[d] - mu) * rs * w[d] + bias[d]);
}

// ---------------------------------------------------------------------------
// Kernel 2: f32 [K][N] -> f16 transposed [N][K] (weights), LDS-tiled
// ---------------------------------------------------------------------------
__global__ __launch_bounds__(256) void mha_cvt_transpose_kernel(
    const float* __restrict__ src, _Float16* __restrict__ dst, int K, int N) {
  __shared__ float tile[32][33];
  const int k0 = blockIdx.y * 32, n0 = blockIdx.x * 32;
  const int tx = threadIdx.x & 31, ty = threadIdx.x >> 5;   // 32 x 8
#pragma unroll
  for (int p = 0; p < 4; ++p)
    tile[ty + p * 8][tx] = src[(size_t)(k0 + ty + p * 8) * N + n0 + tx];
  __syncthreads();
#pragma unroll
  for (int p = 0; p < 4; ++p)
    dst[(size_t)(n0 + ty + p * 8) * K + k0 + tx] =
        (_Float16)tile[tx][ty + p * 8];
}

// ---------------------------------------------------------------------------
// Kernel 3/6: tiled WMMA GEMM  C[M,N](f32) = A[M,K](f16) * B^T[N,K](f16)
// block tile 128x128, 8 waves (4x2), wave tile 32x64 (2x4 WMMA), K-step 32.
// Async double-buffered LDS staging: issue tile k+1 while computing tile k.
// ---------------------------------------------------------------------------
namespace {
constexpr int GBM = 128, GBN = 128, GBK = 32;
constexpr int LDSK = GBK + 16;     // 48 halves = 96 B row stride (16-B aligned)
}

__global__ __launch_bounds__(256) void mha_gemm_kernel(
    const _Float16* __restrict__ A,  // [M][K]
    const _Float16* __restrict__ Bt, // [N][K]
    float* __restrict__ C, int M, int N, int K) {
  __shared__ alignas(16) _Float16 As[2][GBM * LDSK];
  __shared__ alignas(16) _Float16 Bs[2][GBN * LDSK];

  const int tid  = threadIdx.x;
  const int lane = tid & 31;
  const int wave = tid >> 5;
  const int wm = wave >> 1;                 // 0..3 -> rows wm*32
  const int wn = wave & 1;                  // 0..1 -> cols wn*64
  const int row0 = blockIdx.y * GBM;
  const int col0 = blockIdx.x * GBN;

  const int s_row = tid >> 2;               // 0..63 (two passes)
  const int s_cg  = (tid & 3) * 8;          // 16-B column group

  v8f acc[2][4] = {};

  auto stage_tiles = [&](int k0, int buf) {
#pragma unroll
    for (int p = 0; p < 2; ++p) {
      const int r = s_row + p * 64;
      stage16(A + (size_t)(row0 + r) * K + k0 + s_cg,
              &As[buf][r * LDSK + s_cg]);
      stage16(Bt + (size_t)(col0 + r) * K + k0 + s_cg,
              &Bs[buf][r * LDSK + s_cg]);
    }
  };

  stage_tiles(0, 0);
  stage_wait();
  __syncthreads();

  const int nk = K / GBK;
  for (int kk = 0; kk < nk; ++kk) {
    const int buf = kk & 1;
    if (kk + 1 < nk) stage_tiles((kk + 1) * GBK, buf ^ 1);

    v16h af[2], bf[4];
#pragma unroll
    for (int mt = 0; mt < 2; ++mt)
      af[mt] = load_frag_a(&As[buf][(wm * 32 + mt * 16) * LDSK], LDSK);
#pragma unroll
    for (int nt = 0; nt < 4; ++nt)
      bf[nt] = load_frag_b(&Bs[buf][(wn * 64 + nt * 16) * LDSK], LDSK);

#pragma unroll
    for (int mt = 0; mt < 2; ++mt)
#pragma unroll
      for (int nt = 0; nt < 4; ++nt)
        acc[mt][nt] = __builtin_amdgcn_wmma_f32_16x16x32_f16(
            false, af[mt], false, bf[nt], (short)0, acc[mt][nt], false, false);

    stage_wait();
    __syncthreads();
  }

  const int n  = lane & 15;
  const int mb = (lane >> 4) << 3;
#pragma unroll
  for (int mt = 0; mt < 2; ++mt)
#pragma unroll
    for (int nt = 0; nt < 4; ++nt)
#pragma unroll
      for (int r = 0; r < 8; ++r) {
        const int row = row0 + wm * 32 + mt * 16 + mb + r;
        const int col = col0 + wn * 64 + nt * 16 + n;
        C[(size_t)row * N + col] = acc[mt][nt][r];
      }
}

// ---------------------------------------------------------------------------
// Kernel 4: Q/K LayerNorm (scale only) + RoPE + head split, V convert
// ---------------------------------------------------------------------------
__global__ __launch_bounds__(256) void mha_qkv_post_kernel(
    const float* __restrict__ qkv, const float* __restrict__ qw,
    const float* __restrict__ kw, _Float16* __restrict__ Qo,
    _Float16* __restrict__ Ko, _Float16* __restrict__ Vo) {
  __shared__ float  qn[kD];
  __shared__ float  kn[kD];
  __shared__ float4 red[8];
  const int token = blockIdx.x;
  const int b = token / kL, l = token % kL;
  const float* row = qkv + (size_t)token * (3 * kD);

  float sq = 0.f, sqq = 0.f, sk = 0.f, skk = 0.f;
  for (int d = threadIdx.x; d < kD; d += 256) {
    const float qv = row[d];
    const float kv = row[kD + d];
    qn[d] = qv; kn[d] = kv;
    sq += qv; sqq += qv * qv;
    sk += kv; skk += kv * kv;
  }
  const float4 tot = block_reduce4(make_float4(sq, sqq, sk, skk), red);
  const float qmu = tot.x * (1.0f / kD);
  const float qrs = rsqrtf(tot.y * (1.0f / kD) - qmu * qmu + kLnEps);
  const float kmu = tot.z * (1.0f / kD);
  const float krs = rsqrtf(tot.w * (1.0f / kD) - kmu * kmu + kLnEps);
  __syncthreads();
  for (int d = threadIdx.x; d < kD; d += 256) {
    qn[d] = (qn[d] - qmu) * qrs * qw[d];
    kn[d] = (kn[d] - kmu) * krs * kw[d];
  }
  __syncthreads();
  for (int d = threadIdx.x; d < kD; d += 256) {
    const int h = d >> 6, r = d & 63, j = r & 31;
    const float invf = __expf(-(float)(2 * j) * (1.0f / kDh) * kLog10000);
    const float ang = (float)l * invf;
    const float c = __cosf(ang), s = __sinf(ang);
    const float qv = qn[d], kv = kn[d];
    const float qo = (r < 32) ? (qv * c - qn[d + 32] * s)
                              : (qv * c + qn[d - 32] * s);
    const float ko = (r < 32) ? (kv * c - kn[d + 32] * s)
                              : (kv * c + kn[d - 32] * s);
    const size_t oidx = (((size_t)(b * kH + h) * kL) + l) * kDh + r;
    Qo[oidx] = (_Float16)qo;
    Ko[oidx] = (_Float16)ko;
    Vo[oidx] = (_Float16)row[2 * kD + d];
  }
}

// ---------------------------------------------------------------------------
// Kernel 5: flash attention with packed-sequence mask.
// grid = (L/256, B*H); 8 waves/block, each wave owns TWO 16-row Q tiles.
// K staged async [key][dh]; V staged transposed+permuted [dh][key'] with
// packed b64 stores. P packed as half2 (key-pair interleaved contraction).
// ---------------------------------------------------------------------------
namespace {
constexpr int ABK  = 128;            // staged keys per iteration
constexpr int VSTR = ABK + 8;        // 136 halves = 272 B (16-B aligned)
}

__global__ __launch_bounds__(256) void mha_attn_kernel(
    const _Float16* __restrict__ Q, const _Float16* __restrict__ Kk,
    const _Float16* __restrict__ V, const long long* __restrict__ seq,
    _Float16* __restrict__ Ctx) {
  __shared__ alignas(16) _Float16 kS[ABK * kDh];        // [key][dh]
  __shared__ alignas(16) _Float16 vS[kDh * VSTR];       // [dh][key'] permuted
  __shared__ alignas(16) _Float16 pS[8][2][16 * 32];    // per-wave P tiles
  __shared__ int sidS[ABK];

  const int tid  = threadIdx.x;
  const int lane = tid & 31;
  const int wave = tid >> 5;
  const int bh = blockIdx.y;                // b*H + h
  const int b = bh >> 4, h = bh & 15;
  const int q0 = blockIdx.x * 256 + wave * 32;

  const _Float16* qbase = Q + (size_t)bh * kL * kDh;
  const _Float16* kbase = Kk + (size_t)bh * kL * kDh;
  const _Float16* vbase = V + (size_t)bh * kL * kDh;
  const long long* sb = seq + (size_t)b * kL;

  // Q fragments: [q-tile][dh chunk]
  v16h qf[2][2];
#pragma unroll
  for (int qt = 0; qt < 2; ++qt)
#pragma unroll
    for (int c = 0; c < 2; ++c)
      qf[qt][c] = load_frag_a(qbase + (size_t)(q0 + qt * 16) * kDh + c * 32, kDh);

  const int n16 = lane & 15;
  const int mb  = (lane >> 4) << 3;          // this lane's 8 rows
  int   qid[2][8];
  float m_i[2][8], l_i[2][8];
  v8f o[2][4] = {};
#pragma unroll
  for (int qt = 0; qt < 2; ++qt)
#pragma unroll
    for (int r = 0; r < 8; ++r) {
      qid[qt][r] = (int)sb[q0 + qt * 16 + mb + r];
      m_i[qt][r] = -3.0e38f;
      l_i[qt][r] = 0.f;
    }

  // V staging decomposition: thread handles 4 key rows x 8 dh values,
  // writing 8 packed b64 stores (4 permuted-adjacent columns each).
  const int vg   = tid >> 3;                 // 0..31: 4-key group
  const int vdh0 = (tid & 7) * 8;            // dh chunk
  const int vblk = (vg >> 3) * 32;           // 32-key block base
  const int vn0  = (vg & 7) * 2;             // n, n+1 within block

  for (int kv0 = 0; kv0 < kL; kv0 += ABK) {
    // K tile: async b128 copies, [key][dh]
    for (int i = tid; i < ABK * kDh / 8; i += 256)
      stage16(kbase + (size_t)kv0 * kDh + i * 8, kS + i * 8);
    // V tile: registers -> transposed/permuted LDS, packed b64 stores
    {
      const _Float16* vsrc = vbase + (size_t)(kv0 + vblk) * kDh + vdh0;
      uint4 rv[4];
      rv[0] = *(const uint4*)(vsrc + (size_t)(vn0)      * kDh);
      rv[1] = *(const uint4*)(vsrc + (size_t)(vn0 + 16) * kDh);
      rv[2] = *(const uint4*)(vsrc + (size_t)(vn0 + 1)  * kDh);
      rv[3] = *(const uint4*)(vsrc + (size_t)(vn0 + 17) * kDh);
      const _Float16* h0 = (const _Float16*)&rv[0];
      const _Float16* h1 = (const _Float16*)&rv[1];
      const _Float16* h2 = (const _Float16*)&rv[2];
      const _Float16* h3 = (const _Float16*)&rv[3];
      const int col0 = vblk + 2 * vn0;
#pragma unroll
      for (int e = 0; e < 8; ++e) {
        union { _Float16 hx[4]; unsigned long long u; } pk;
        pk.hx[0] = h0[e]; pk.hx[1] = h1[e];
        pk.hx[2] = h2[e]; pk.hx[3] = h3[e];
        *(unsigned long long*)&vS[(size_t)(vdh0 + e) * VSTR + col0] = pk.u;
      }
    }
    if (tid < ABK) sidS[tid] = (int)sb[kv0 + tid];
    stage_wait();
    __syncthreads();

#pragma unroll
    for (int sub = 0; sub < ABK / 32; ++sub) {
      const int key0 = sub * 32;
      // score B-frags, shared by both q tiles: [half][dh chunk]
      v16h sb0[2], sb1[2];
#pragma unroll
      for (int half = 0; half < 2; ++half) {
        const _Float16* kt = kS + (size_t)(key0 + half * 16) * kDh;
        sb0[half] = load_frag_b(kt, kDh);        // dh 0..31
        sb1[half] = load_frag_b(kt + 32, kDh);   // dh 32..63
      }
#pragma unroll
      for (int qt = 0; qt < 2; ++qt) {
        v8f sc[2] = {};
#pragma unroll
        for (int half = 0; half < 2; ++half) {
          sc[half] = __builtin_amdgcn_wmma_f32_16x16x32_f16(
              false, qf[qt][0], false, sb0[half], (short)0, sc[half], false, false);
          sc[half] = __builtin_amdgcn_wmma_f32_16x16x32_f16(
              false, qf[qt][1], false, sb1[half], (short)0, sc[half], false, false);
        }
        // mask + online softmax; rows live across 16-lane half-waves
        float alpha[8];
#pragma unroll
        for (int r = 0; r < 8; ++r) {
          const bool ok0 = (sidS[key0 + n16] == qid[qt][r]);
          const bool ok1 = (sidS[key0 + 16 + n16] == qid[qt][r]);
          const float s0 = ok0 ? sc[0][r] * 0.125f : -1.0e30f;
          const float s1 = ok1 ? sc[1][r] * 0.125f : -1.0e30f;
          float rmax = fmaxf(s0, s1);
#pragma unroll
          for (int msk = 1; msk < 16; msk <<= 1)
            rmax = fmaxf(rmax, __shfl_xor(rmax, msk, 32));
          const float mnew = fmaxf(m_i[qt][r], rmax);
          alpha[r] = __expf(m_i[qt][r] - mnew);
          m_i[qt][r] = mnew;
          const float p0 = __expf(s0 - mnew);
          const float p1 = __expf(s1 - mnew);
          float rsum = p0 + p1;
#pragma unroll
          for (int msk = 1; msk < 16; msk <<= 1)
            rsum += __shfl_xor(rsum, msk, 32);
          l_i[qt][r] = l_i[qt][r] * alpha[r] + rsum;
          // packed half2 store, contraction index c = 2*n16 (+1 for key+16)
          union { _Float16 hx[2]; unsigned u; } pk;
          pk.hx[0] = (_Float16)p0;
          pk.hx[1] = (_Float16)p1;
          *(unsigned*)&pS[wave][qt][(mb + r) * 32 + 2 * n16] = pk.u;
        }
#pragma unroll
        for (int t = 0; t < 4; ++t)
#pragma unroll
          for (int r = 0; r < 8; ++r) o[qt][t][r] *= alpha[r];
      }
      __syncthreads();                       // pS stores -> A-frag reloads
      // V B-frags (permuted key order matches P packing), shared by q tiles
      v16h vf[4];
#pragma unroll
      for (int t = 0; t < 4; ++t)
        vf[t] = load_frag_b(vS + (size_t)(t * 16) * VSTR + key0, VSTR);
#pragma unroll
      for (int qt = 0; qt < 2; ++qt) {
        const v16h pf = load_frag_a(&pS[wave][qt][0], 32);
#pragma unroll
        for (int t = 0; t < 4; ++t)
          o[qt][t] = __builtin_amdgcn_wmma_f32_16x16x32_f16(
              false, pf, false, vf[t], (short)0, o[qt][t], false, false);
      }
      __syncthreads();
    }
    __syncthreads();
  }

  // finalize: ctx[b, l, h, dh] (flattens to [8192,1024] for out-proj GEMM)
#pragma unroll
  for (int qt = 0; qt < 2; ++qt)
#pragma unroll
    for (int t = 0; t < 4; ++t)
#pragma unroll
      for (int r = 0; r < 8; ++r) {
        const size_t idx =
            ((size_t)(b * kL + q0 + qt * 16 + mb + r) * kD) + h * kDh + t * 16 + n16;
        Ctx[idx] = (_Float16)(o[qt][t][r] / l_i[qt][r]);
      }
}

// ---------------------------------------------------------------------------
// Host-side orchestration
// ---------------------------------------------------------------------------
extern "C" void kernel_launch(void* const* d_in, const int* in_sizes, int n_in,
                              void* d_out, int out_size, void* d_ws, size_t ws_size,
                              hipStream_t stream) {
  (void)in_sizes; (void)n_in; (void)out_size; (void)ws_size;
  const float*     x     = (const float*)d_in[0];
  const long long* seqid = (const long long*)d_in[1];   // int64 sequence ids
  const float*     ln_w  = (const float*)d_in[2];
  const float*     ln_b  = (const float*)d_in[3];
  const float*     w_qkv = (const float*)d_in[4];
  const float*     q_ln  = (const float*)d_in[5];
  const float*     k_ln  = (const float*)d_in[6];
  const float*     w_out = (const float*)d_in[7];
  float* out = (float*)d_out;

  char* ws = (char*)d_ws;
  const size_t sz_xn   = (size_t)kBL * kD * sizeof(_Float16);        // 16 MB
  const size_t sz_wqkv = (size_t)kD * 3 * kD * sizeof(_Float16);     //  6 MB
  const size_t sz_wout = (size_t)kD * kD * sizeof(_Float16);         //  2 MB
  const size_t sz_qkv  = (size_t)kBL * 3 * kD * sizeof(float);       // 96 MB

  _Float16* xn     = (_Float16*)(ws);
  _Float16* wqkvT  = (_Float16*)(ws + sz_xn);                        // [3D][D]
  _Float16* woutT  = (_Float16*)(ws + sz_xn + sz_wqkv);              // [D][D]
  float*    qkv    = (float*)   (ws + sz_xn + sz_wqkv + sz_wout);
  _Float16* qh     = (_Float16*)(ws + sz_xn + sz_wqkv + sz_wout + sz_qkv);
  _Float16* kh     = qh + (size_t)kBL * kD;
  _Float16* vh     = kh + (size_t)kBL * kD;
  _Float16* ctx    = xn;   // xn dead after GEMM1; reuse for ctx

  // 1) LN(x) -> f16
  mha_ln_x_kernel<<<kBL, 256, 0, stream>>>(x, ln_w, ln_b, xn);
  // 2) weight convert+transpose to [N][K] f16
  mha_cvt_transpose_kernel<<<dim3(3 * kD / 32, kD / 32), 256, 0, stream>>>(
      w_qkv, wqkvT, kD, 3 * kD);
  mha_cvt_transpose_kernel<<<dim3(kD / 32, kD / 32), 256, 0, stream>>>(
      w_out, woutT, kD, kD);
  // 3) QKV GEMM: [8192,1024] x [1024,3072]
  mha_gemm_kernel<<<dim3(3 * kD / GBN, kBL / GBM), 256, 0, stream>>>(
      xn, wqkvT, qkv, kBL, 3 * kD, kD);
  // 4) qk-LN + RoPE + head split
  mha_qkv_post_kernel<<<kBL, 256, 0, stream>>>(qkv, q_ln, k_ln, qh, kh, vh);
  // 5) flash attention
  mha_attn_kernel<<<dim3(kL / 256, kB * kH), 256, 0, stream>>>(
      qh, kh, vh, seqid, ctx);
  // 6) out projection: [8192,1024] x [1024,1024]
  mha_gemm_kernel<<<dim3(kD / GBN, kBL / GBM), 256, 0, stream>>>(
      ctx, woutT, out, kBL, kD, kD);
}
